// rnnLU_55173149884953
// MI455X (gfx1250) — matrix operand order repaired and checked by reading, not compiled, not run on hardware
//
#include <hip/hip_runtime.h>
#include <math.h>

typedef __attribute__((ext_vector_type(16))) _Float16 v16h;
typedef __attribute__((ext_vector_type(8)))  float    v8f;
typedef __attribute__((ext_vector_type(4)))  float    v4f;

#define NN 1024   // hidden units
#define MM 8      // outputs
#define TT 8192   // time steps

// ---------------------------------------------------------------------------
// Prep kernel: Jeff = J + Wfb @ W, converted to f16 and packed into this code's contiguous-K
// per-lane A-fragment layout for V_WMMA_F32_16X16X32_F16 (B uses the same K order):
//   lane l in 0..15  -> row m = tm*16 + l,  K order: 0..15
//   lane l in 16..31 -> row m = tm*16+l-16, K order: 16..31
// Flat packed index: (((tm*32 + tk)*32 + lane)*16 + e)  -> 2 MB in d_ws.
// ---------------------------------------------------------------------------
__global__ void rnn_prep_pack(const float* __restrict__ J,
                              const float* __restrict__ W,
                              const float* __restrict__ Wfb,
                              _Float16* __restrict__ Apack) {
  int p = blockIdx.x * blockDim.x + threadIdx.x;
  if (p >= (NN / 16) * (NN / 32) * 32 * 16) return;
  int e    =  p        & 15;
  int lane = (p >> 4)  & 31;
  int tk   = (p >> 9)  & 31;
  int tm   =  p >> 14;
  int m    = tm * 16 + (lane & 15);
  int half = lane >> 4;
  int kl   = e + (half << 4);  // {0..15} / {16..31}
  int k    = tk * 32 + kl;
  float v = J[m * NN + k];
#pragma unroll
  for (int c = 0; c < MM; ++c)
    v += Wfb[m * MM + c] * W[c * NN + k];   // fold feedback: Jeff = J + Wfb@W
  Apack[p] = (_Float16)v;
}

// ---------------------------------------------------------------------------
// Persistent single-workgroup recurrence: 32 waves, each owns 32 rows
// (2 WMMA row tiles). Per step, 2048 v_wmma_f32_16x16x32_f16 total.
// ---------------------------------------------------------------------------
__global__ __launch_bounds__(1024)
void rnn_recur(const _Float16* __restrict__ Apack,
               const float*    __restrict__ inp,   // [N][T]
               const float*    __restrict__ W,     // [M][N]
               float*          __restrict__ pred)  // [M][T]
{
  __shared__ __align__(32) _Float16 r_h[NN];  // f16 r for WMMA B fragments
  __shared__ float r_f[NN];                   // f32 r for W@r readout
  __shared__ __align__(16) float ybuf[NN];    // Jeff@r result

  const int tid  = threadIdx.x;
  const int lane = tid & 31;
  const int wave = tid >> 5;            // 0..31
  const int row0 = wave * 32;           // this wave's rows: tiles 2w, 2w+1

  // init: r_0 = 0, pred[:,0] = 0
  if (tid < MM) pred[tid * TT] = 0.0f;
  r_h[tid] = (_Float16)0.0f;
  r_f[tid] = 0.0f;
  __syncthreads();

  const _Float16* A0 = Apack + (size_t)((2 * wave    ) * 32) * 512 + lane * 16;
  const _Float16* A1 = Apack + (size_t)((2 * wave + 1) * 32) * 512 + lane * 16;
  const int bofs = (lane & 16);         // lanes 0-15 take K 0..15, 16-31 take K 16..31

  for (int t = 1; t < TT; ++t) {
    // issue the (uncoalesced but WGP$-resident) input load early; its latency
    // hides behind the 64-WMMA accumulation below
    float inpv = inp[(size_t)tid * TT + t];

    v8f acc0 = {};
    v8f acc1 = {};
#pragma unroll 4
    for (int tk = 0; tk < 32; ++tk) {
      v16h bf = *(const v16h*)&r_h[tk * 32 + bofs];          // ds_load, broadcast cols
      v16h a0 = *(const v16h*)(A0 + tk * 512);               // global_load 32B/lane
      v16h a1 = *(const v16h*)(A1 + tk * 512);
      acc0 = __builtin_amdgcn_wmma_f32_16x16x32_f16(false, a0, false, bf,
                                                    (short)0, acc0, false, false);
      acc1 = __builtin_amdgcn_wmma_f32_16x16x32_f16(false, a1, false, bf,
                                                    (short)0, acc1, false, false);
    }
    // D layout: VGPR v -> M=v (lanes 0-15) / M=v+8 (lanes 16-31); cols identical.
    // Accumulator VGPRs are consecutive -> write as b128 vector stores.
    if (lane == 0) {
      *(v4f*)&ybuf[row0     ] = acc0.lo;
      *(v4f*)&ybuf[row0 +  4] = acc0.hi;
      *(v4f*)&ybuf[row0 + 16] = acc1.lo;
      *(v4f*)&ybuf[row0 + 20] = acc1.hi;
    } else if (lane == 16) {
      *(v4f*)&ybuf[row0 +  8] = acc0.lo;
      *(v4f*)&ybuf[row0 + 12] = acc0.hi;
      *(v4f*)&ybuf[row0 + 24] = acc1.lo;
      *(v4f*)&ybuf[row0 + 28] = acc1.hi;
    }
    __syncthreads();                    // all waves done reading old r_h

    // x = Jeff@r + input[:,t];  r = tanh(x)
    float x = ybuf[tid] + inpv;
    float r = tanhf(x);
    r_f[tid] = r;
    r_h[tid] = (_Float16)r;
    __syncthreads();                    // new r published

    // pred[:,t] = W @ r   (waves 0..7, wave32 shuffle reduction)
    if (wave < MM) {
      const float* wr = W + wave * NN;
      float s = 0.0f;
#pragma unroll 8
      for (int i = lane; i < NN; i += 32) s += wr[i] * r_f[i];
#pragma unroll
      for (int off = 16; off; off >>= 1) s += __shfl_xor(s, off, 32);
      if (lane == 0) pred[wave * TT + t] = s;
    }
  }
}

// ---------------------------------------------------------------------------
extern "C" void kernel_launch(void* const* d_in, const int* in_sizes, int n_in,
                              void* d_out, int out_size, void* d_ws, size_t ws_size,
                              hipStream_t stream) {
  const float* inp = (const float*)d_in[0];   // (N, T)
  const float* J   = (const float*)d_in[1];   // (N, N)
  const float* W   = (const float*)d_in[2];   // (M, N)
  const float* Wfb = (const float*)d_in[3];   // (N, M)
  float* pred      = (float*)d_out;           // (M, T)
  _Float16* Apack  = (_Float16*)d_ws;         // 2 MB packed Jeff (f16)

  (void)in_sizes; (void)n_in; (void)out_size; (void)ws_size;

  const int total = (NN / 16) * (NN / 32) * 32 * 16;   // 1M packed elements
  rnn_prep_pack<<<(total + 255) / 256, 256, 0, stream>>>(J, W, Wfb, Apack);
  rnn_recur<<<1, 1024, 0, stream>>>(Apack, inp, W, pred);
}
